// ScaledDotProduct_25778393711010
// MI455X (gfx1250) — matrix operand order repaired and checked by reading, not compiled, not run on hardware
//
#include <hip/hip_runtime.h>

typedef __attribute__((ext_vector_type(16))) _Float16 v16h;
typedef __attribute__((ext_vector_type(8)))  _Float16 v8h;
typedef __attribute__((ext_vector_type(4)))  _Float16 v4h;
typedef __attribute__((ext_vector_type(2)))  _Float16 v2h;
typedef __attribute__((ext_vector_type(8)))  float    v8f;

#define S_LEN 2048
#define D_DIM 64
#define BM    128  // query rows per block (8 waves x 16)
#define BN    64   // keys per iteration
#define NWAVE 8

// packed f32 -> f16x2 convert (v_cvt_pk_f16_f32), re-typed to _Float16 vector
__device__ __forceinline__ v2h pkh(float a, float b) {
    return __builtin_bit_cast(v2h, __builtin_amdgcn_cvt_pkrtz(a, b));
}

// ds_swizzle xor-mode butterfly (group-of-32: and=0x1f, or=0, xor=IMM>>10)
template <int IMM>
__device__ __forceinline__ float swz(float x) {
    return __int_as_float(__builtin_amdgcn_ds_swizzle(__float_as_int(x), IMM));
}
__device__ __forceinline__ float redMax16(float v) {
    v = fmaxf(v, swz<0x041f>(v));   // xor 1
    v = fmaxf(v, swz<0x081f>(v));   // xor 2
    v = fmaxf(v, swz<0x101f>(v));   // xor 4
    v = fmaxf(v, swz<0x201f>(v));   // xor 8
    return v;
}
__device__ __forceinline__ float redSum16(float v) {
    v += swz<0x041f>(v);
    v += swz<0x081f>(v);
    v += swz<0x101f>(v);
    v += swz<0x201f>(v);
    return v;
}

__global__ __launch_bounds__(256)
void fa_fwd_causal_f16wmma(const float* __restrict__ Q,
                           const float* __restrict__ K,
                           const float* __restrict__ V,
                           float* __restrict__ O)
{
    __shared__ __align__(16) _Float16 ldsK [BN * D_DIM];        // [key][d]      8KB
    __shared__ __align__(16) _Float16 ldsVt[D_DIM * BN];        // [d][key]      8KB
    __shared__ __align__(16) _Float16 ldsP [NWAVE * 16 * BN];   // per-wave P   16KB

    const int tid  = threadIdx.x;
    const int wave = tid >> 5;
    const int lane = tid & 31;
    const int lh   = lane & 15;
    const int lg   = lane >> 4;

    const int    qBase = blockIdx.x * BM;
    const size_t bh    = blockIdx.y;
    const float* Qb = Q + bh * (size_t)(S_LEN * D_DIM);
    const float* Kb = K + bh * (size_t)(S_LEN * D_DIM);
    const float* Vb = V + bh * (size_t)(S_LEN * D_DIM);
    float*       Ob = O + bh * (size_t)(S_LEN * D_DIM);

    const int qRow0 = qBase + wave * 16;
    const int qLast = qRow0 + 15;

    // exp2-domain scale: (1/sqrt(64)) * log2(e), folded into the Q fragment
    const float kscale = 0.125f * 1.44269504088896341f;

    // ---- Q A-fragments (16x32 f16), d-chunks [0,32) and [32,64), pre-scaled ----
    // A layout: lane holds row m = lane&15; half h -> K = h + 16*lg
    v16h aQ0, aQ1;
    {
        const float* qrow = Qb + (size_t)(qRow0 + lh) * D_DIM;
        #pragma unroll
        for (int h = 0; h < 16; ++h) {
            const int kk = h + 16 * lg;
            aQ0[h] = (_Float16)(qrow[kk]      * kscale);
            aQ1[h] = (_Float16)(qrow[32 + kk] * kscale);
        }
    }

    v8f o0 = {}, o1 = {}, o2 = {}, o3 = {};
    float mrow[8], lrow[8];
    #pragma unroll
    for (int r = 0; r < 8; ++r) { mrow[r] = -1.0e30f; lrow[r] = 0.0f; }

    const int kvEnd = qBase + BM;

    for (int kv0 = 0; kv0 < kvEnd; kv0 += BN) {
        // ---- cooperative K tile: f32 -> packed f16, b64 stores ----
        // 64 rows x 16 float4 = 1024 tasks / 256 threads = 4 each
        #pragma unroll
        for (int it = 0; it < 4; ++it) {
            const int i   = tid + it * 256;
            const int row = i >> 4;
            const int c4  = i & 15;
            const float4 k4 = *(const float4*)(Kb + (size_t)(kv0 + row) * D_DIM + c4 * 4);
            const v2h p01 = pkh(k4.x, k4.y);
            const v2h p23 = pkh(k4.z, k4.w);
            v4h pk; pk[0] = p01[0]; pk[1] = p01[1]; pk[2] = p23[0]; pk[3] = p23[1];
            *(v4h*)&ldsK[row * D_DIM + c4 * 4] = pk;
        }
        // ---- cooperative V tile (transposed): row pairs -> packed b32 stores ----
        // 32 row-pairs x 16 float4 = 512 tasks / 256 threads = 2 each
        #pragma unroll
        for (int it = 0; it < 2; ++it) {
            const int i  = tid + it * 256;
            const int p  = i >> 4;        // row pair index (0..31)
            const int c4 = i & 15;
            const float4 a = *(const float4*)(Vb + (size_t)(kv0 + 2 * p)     * D_DIM + c4 * 4);
            const float4 b = *(const float4*)(Vb + (size_t)(kv0 + 2 * p + 1) * D_DIM + c4 * 4);
            *(v2h*)&ldsVt[(c4 * 4 + 0) * BN + 2 * p] = pkh(a.x, b.x);
            *(v2h*)&ldsVt[(c4 * 4 + 1) * BN + 2 * p] = pkh(a.y, b.y);
            *(v2h*)&ldsVt[(c4 * 4 + 2) * BN + 2 * p] = pkh(a.z, b.z);
            *(v2h*)&ldsVt[(c4 * 4 + 3) * BN + 2 * p] = pkh(a.w, b.w);
        }
        __syncthreads();

        // ---- prefetch next KV tile into cache (global_prefetch_b8) ----
        const int nkv = kv0 + BN;
        if (nkv < kvEnd) {
            const int prow = tid >> 2;
            const int pcol = (tid & 3) * 16;
            __builtin_prefetch(Kb + (size_t)(nkv + prow) * D_DIM + pcol, 0, 0);
            __builtin_prefetch(Vb + (size_t)(nkv + prow) * D_DIM + pcol, 0, 0);
        }

        if (kv0 <= qLast) {  // wave-uniform: at least one unmasked key this block
            // ---- S = (scaled Q) K^T : up to four 16x16 tiles over 64 keys ----
            v8f s[4];
            bool live[4];
            #pragma unroll
            for (int nt = 0; nt < 4; ++nt) {
                live[nt] = (kv0 + nt * 16) <= qLast;        // wave-uniform
                if (live[nt]) {
                    const int j = nt * 16 + lh;             // B-layout: N = lh
                    const v16h b0 = *(const v16h*)&ldsK[j * D_DIM +  0 + 16 * lg];
                    const v16h b1 = *(const v16h*)&ldsK[j * D_DIM + 32 + 16 * lg];
                    v8f z = {};
                    v8f acc = __builtin_amdgcn_wmma_f32_16x16x32_f16(
                                  false, aQ0, false, b0, (short)0, z, false, false);
                    acc     = __builtin_amdgcn_wmma_f32_16x16x32_f16(
                                  false, aQ1, false, b1, (short)0, acc, false, false);
                    s[nt] = acc;
                    // causal mask: only the diagonal tile ever needs it (uniform)
                    if ((kv0 + nt * 16 + 15) > qRow0) {
                        const int key = kv0 + nt * 16 + lh;
                        #pragma unroll
                        for (int r = 0; r < 8; ++r)
                            if (key > (qRow0 + r + 8 * lg)) s[nt][r] = -1.0e30f;
                    }
                }
            }

            // ---- online softmax (exp2 domain); reductions via ds_swizzle ----
            #pragma unroll
            for (int r = 0; r < 8; ++r) {
                float rm = -1.0e30f;
                #pragma unroll
                for (int nt = 0; nt < 4; ++nt)
                    if (live[nt]) rm = fmaxf(rm, s[nt][r]);
                rm = redMax16(rm);
                const float mNew  = fmaxf(mrow[r], rm);
                const float alpha = __builtin_amdgcn_exp2f(mrow[r] - mNew);
                float rs = 0.0f;
                #pragma unroll
                for (int nt = 0; nt < 4; ++nt) {
                    if (live[nt]) {
                        const float e = __builtin_amdgcn_exp2f(s[nt][r] - mNew);
                        s[nt][r] = e;
                        rs += e;
                    }
                }
                rs = redSum16(rs);
                lrow[r] = lrow[r] * alpha + rs;
                mrow[r] = mNew;
                o0[r] *= alpha; o1[r] *= alpha; o2[r] *= alpha; o3[r] *= alpha;
            }

            // ---- P: C-layout -> LDS (uniform branch per tile; zeros if dead) ----
            _Float16* pw = &ldsP[wave * 16 * BN];
            #pragma unroll
            for (int nt = 0; nt < 4; ++nt) {
                if (live[nt]) {
                    #pragma unroll
                    for (int r = 0; r < 8; ++r)
                        pw[(r + 8 * lg) * BN + nt * 16 + lh] = (_Float16)s[nt][r];
                } else {
                    #pragma unroll
                    for (int r = 0; r < 8; ++r)
                        pw[(r + 8 * lg) * BN + nt * 16 + lh] = (_Float16)0.0f;
                }
            }

            // ---- reload P as A-fragments (two K=32 chunks) ----
            v16h aP0, aP1;
            {
                const v8h l0 = *(const v8h*)&pw[lh * BN +  0 + 16 * lg];
                const v8h h0 = *(const v8h*)&pw[lh * BN +  8 + 16 * lg];
                const v8h l1 = *(const v8h*)&pw[lh * BN + 32 + 16 * lg];
                const v8h h1 = *(const v8h*)&pw[lh * BN + 40 + 16 * lg];
                #pragma unroll
                for (int h = 0; h < 8; ++h) {
                    aP0[h] = l0[h]; aP0[h + 8] = h0[h];
                    aP1[h] = l1[h]; aP1[h + 8] = h1[h];
                }
            }

            // ---- O += P V : keys chunk 0 (0..31) always, chunk 1 if live ----
            {
                const v16h bV0 = *(const v16h*)&ldsVt[( 0 + lh) * BN +  0 + 16 * lg];
                const v16h bV1 = *(const v16h*)&ldsVt[(16 + lh) * BN +  0 + 16 * lg];
                const v16h bV2 = *(const v16h*)&ldsVt[(32 + lh) * BN +  0 + 16 * lg];
                const v16h bV3 = *(const v16h*)&ldsVt[(48 + lh) * BN +  0 + 16 * lg];
                o0 = __builtin_amdgcn_wmma_f32_16x16x32_f16(false, aP0, false, bV0, (short)0, o0, false, false);
                o1 = __builtin_amdgcn_wmma_f32_16x16x32_f16(false, aP0, false, bV1, (short)0, o1, false, false);
                o2 = __builtin_amdgcn_wmma_f32_16x16x32_f16(false, aP0, false, bV2, (short)0, o2, false, false);
                o3 = __builtin_amdgcn_wmma_f32_16x16x32_f16(false, aP0, false, bV3, (short)0, o3, false, false);
            }
            if (live[2]) {  // any key in 32..63 unmasked (wave-uniform)
                const v16h bV0 = *(const v16h*)&ldsVt[( 0 + lh) * BN + 32 + 16 * lg];
                const v16h bV1 = *(const v16h*)&ldsVt[(16 + lh) * BN + 32 + 16 * lg];
                const v16h bV2 = *(const v16h*)&ldsVt[(32 + lh) * BN + 32 + 16 * lg];
                const v16h bV3 = *(const v16h*)&ldsVt[(48 + lh) * BN + 32 + 16 * lg];
                o0 = __builtin_amdgcn_wmma_f32_16x16x32_f16(false, aP1, false, bV0, (short)0, o0, false, false);
                o1 = __builtin_amdgcn_wmma_f32_16x16x32_f16(false, aP1, false, bV1, (short)0, o1, false, false);
                o2 = __builtin_amdgcn_wmma_f32_16x16x32_f16(false, aP1, false, bV2, (short)0, o2, false, false);
                o3 = __builtin_amdgcn_wmma_f32_16x16x32_f16(false, aP1, false, bV3, (short)0, o3, false, false);
            }
        }
        __syncthreads();
    }

    // ---- epilogue: O = acc / l ----
    #pragma unroll
    for (int r = 0; r < 8; ++r) {
        const int   qi  = qRow0 + r + 8 * lg;
        const float inv = 1.0f / lrow[r];
        float* orow = Ob + (size_t)qi * D_DIM;
        orow[ 0 + lh] = o0[r] * inv;
        orow[16 + lh] = o1[r] * inv;
        orow[32 + lh] = o2[r] * inv;
        orow[48 + lh] = o3[r] * inv;
    }
}

extern "C" void kernel_launch(void* const* d_in, const int* in_sizes, int n_in,
                              void* d_out, int out_size, void* d_ws, size_t ws_size,
                              hipStream_t stream) {
    (void)in_sizes; (void)n_in; (void)out_size; (void)d_ws; (void)ws_size;
    const float* q = (const float*)d_in[0];
    const float* k = (const float*)d_in[1];
    const float* v = (const float*)d_in[2];
    // d_in[3] is the causal mask; causality is applied analytically in-kernel.
    float* o = (float*)d_out;

    dim3 grid(S_LEN / BM, 4 * 16 /* B*H */);
    dim3 block(256);
    fa_fwd_causal_f16wmma<<<grid, block, 0, stream>>>(q, k, v, o);
}
